// Net_43404939493902
// MI455X (gfx1250) — compile-verified, matrix operand order checked by reading, not run on hardware
//
#include <hip/hip_runtime.h>
#include <hip/hip_bf16.h>

#define GAT_N 100000
#define GAT_E 1600000

typedef __attribute__((ext_vector_type(16))) _Float16 v16h;
typedef __attribute__((ext_vector_type(8)))  float    v8f;

// ---------------------------------------------------------------------------
// Operand packing: convert f32 matrices into f16 WMMA fragment layout
// (CDNA5 ISA 7.12.2), padded with zeros to 16x32 / 32x16 tile boundaries.
// A-fragment (16x32, MxK): lane l holds row M=l&15; k index for f16 slot j:
//   v = j>>1, k = ((v>>2)<<4) + ((l>>4)<<3) + ((v&3)<<1) + (j&1)
// B-fragment (32x16, KxN): lane l holds column N=l&15; same k enumeration.
// Packed element order: [tile][lane][j] contiguous -> wave loads 32B/lane.
// ---------------------------------------------------------------------------
__device__ __forceinline__ int frag_k(int lane, int j) {
    int v = j >> 1;
    return ((v >> 2) << 4) + ((lane >> 4) << 3) + ((v & 3) << 1) + (j & 1);
}

__global__ void pack_a_k(const float* __restrict__ A, _Float16* __restrict__ Ap,
                         int M, int K, int Mt, int Kt)
{
    long i = (long)blockIdx.x * blockDim.x + threadIdx.x;
    long total = (long)Mt * Kt * 512;
    if (i >= total) return;
    int  j    = (int)(i & 15);
    int  lane = (int)((i >> 4) & 31);
    long tile = i >> 9;
    int  kt = (int)(tile % Kt);
    int  mt = (int)(tile / Kt);
    int  gm = mt * 16 + (lane & 15);
    int  gk = kt * 32 + frag_k(lane, j);
    float val = (gm < M && gk < K) ? A[(size_t)gm * K + gk] : 0.0f;
    Ap[i] = (_Float16)val;
}

__global__ void pack_b_k(const float* __restrict__ B, _Float16* __restrict__ Bp,
                         int K, int Nc, int Kt, int Nt)
{
    long i = (long)blockIdx.x * blockDim.x + threadIdx.x;
    long total = (long)Kt * Nt * 512;
    if (i >= total) return;
    int  j    = (int)(i & 15);
    int  lane = (int)((i >> 4) & 31);
    long tile = i >> 9;
    int  nt = (int)(tile % Nt);
    int  kt = (int)(tile / Nt);
    int  gk = kt * 32 + frag_k(lane, j);
    int  gn = nt * 16 + (lane & 15);
    float val = (gk < K && gn < Nc) ? B[(size_t)gk * Nc + gn] : 0.0f;
    Bp[i] = (_Float16)val;
}

// ---------------------------------------------------------------------------
// WMMA GEMM over packed fragments. KT = 2 (all layers have K <= 64).
// 4 waves / block, each wave owns one 16x16 output tile (uniform exit keeps
// EXEC all-ones for live waves, as WMMA requires).
// Inner loop: 2x 32B fragment loads + v_wmma, fully unrolled.
// ---------------------------------------------------------------------------
#define KT 2

__global__ void __launch_bounds__(128)
gemm_wmma_packed(const v16h* __restrict__ Ap, const v16h* __restrict__ Bp,
                 const float* __restrict__ bias, float* __restrict__ C,
                 int Mt, int Nt, int M, int Nc)
{
    const int lane = threadIdx.x;                 // 0..31
    const int mt = blockIdx.x * 4 + threadIdx.y;  // wave-uniform
    const int nt = blockIdx.y;
    if (mt >= Mt) return;

    v8f acc = {};
    #pragma unroll
    for (int kt = 0; kt < KT; ++kt) {
        v16h a = Ap[((size_t)mt * KT + kt) * 32 + lane];
        v16h b = Bp[((size_t)kt * Nt + nt) * 32 + lane];
        acc = __builtin_amdgcn_wmma_f32_16x16x32_f16(
                  false, a, false, b, (short)0, acc, false, false);
    }

    // C/D layout: VGPR r -> (M = r + 8*(lane>>4), N = lane&15)
    const int nc   = nt * 16 + (lane & 15);
    const int mrow = mt * 16 + (lane >> 4) * 8;
    if (nc < Nc) {
        float bv = bias ? bias[nc] : 0.0f;
        #pragma unroll
        for (int r = 0; r < 8; ++r) {
            int mm = mrow + r;
            if (mm < M) C[(size_t)mm * Nc + nc] = acc[r] + bv;
        }
    }
}

// ---------------------------------------------------------------------------
// Elementwise / edge kernels
// ---------------------------------------------------------------------------
__device__ __forceinline__ unsigned f2ord(float f) {
    unsigned u = __float_as_uint(f);
    return (u & 0x80000000u) ? ~u : (u | 0x80000000u);
}
__device__ __forceinline__ float ord2f(unsigned u) {
    return __uint_as_float((u & 0x80000000u) ? (u & 0x7FFFFFFFu) : ~u);
}

__global__ void fill_u32(unsigned* __restrict__ p, unsigned v, long n) {
    long i = (long)blockIdx.x * blockDim.x + threadIdx.x;
    if (i < n) p[i] = v;
}

// a_src[n,h] = <xw[n,h,:], att_src[h,:]>, likewise a_dst
__global__ void attn_scores(const float* __restrict__ xw,
                            const float* __restrict__ att_s,
                            const float* __restrict__ att_d,
                            float* __restrict__ a_src, float* __restrict__ a_dst,
                            int n, int H, int D)
{
    long i = (long)blockIdx.x * blockDim.x + threadIdx.x;
    if (i >= (long)n * H) return;
    int node = (int)(i / H), h = (int)(i % H);
    const float* v  = xw + (size_t)node * H * D + (size_t)h * D;
    const float* as = att_s + (size_t)h * D;
    const float* ad = att_d + (size_t)h * D;
    float s1 = 0.f, s2 = 0.f;
    for (int d = 0; d < D; ++d) { s1 += v[d] * as[d]; s2 += v[d] * ad[d]; }
    a_src[i] = s1;
    a_dst[i] = s2;
}

__device__ __forceinline__ void edge_ids(long e, const int* src, const int* dst,
                                         int E_, int& s, int& t)
{
    if (e < E_) { s = src[e]; t = dst[e]; }
    else        { s = (int)(e - E_); t = s; }   // self loops
}

// pass 1: segment max of leaky_relu scores over destination
__global__ void edge_max_k(const int* __restrict__ src, const int* __restrict__ dst,
                           int E_, int n,
                           const float* __restrict__ a_src,
                           const float* __restrict__ a_dst,
                           unsigned* __restrict__ emax, int H)
{
    long e = (long)blockIdx.x * blockDim.x + threadIdx.x;
    if (e >= (long)E_ + n) return;
    int s, t; edge_ids(e, src, dst, E_, s, t);
    for (int h = 0; h < H; ++h) {
        float sc = a_src[(size_t)s * H + h] + a_dst[(size_t)t * H + h];
        sc = sc > 0.f ? sc : 0.2f * sc;
        atomicMax(&emax[(size_t)t * H + h], f2ord(sc));
    }
}

// pass 2: ex = exp(score - max); store it; denom[dst,h] += ex
__global__ void edge_expsum_k(const int* __restrict__ src, const int* __restrict__ dst,
                              int E_, int n,
                              const float* __restrict__ a_src,
                              const float* __restrict__ a_dst,
                              const unsigned* __restrict__ emax,
                              float* __restrict__ ealpha,
                              float* __restrict__ denom, int H)
{
    long e = (long)blockIdx.x * blockDim.x + threadIdx.x;
    if (e >= (long)E_ + n) return;
    int s, t; edge_ids(e, src, dst, E_, s, t);
    for (int h = 0; h < H; ++h) {
        float sc = a_src[(size_t)s * H + h] + a_dst[(size_t)t * H + h];
        sc = sc > 0.f ? sc : 0.2f * sc;
        float ex = __expf(sc - ord2f(emax[(size_t)t * H + h]));
        ealpha[e * H + h] = ex;
        atomicAdd(&denom[(size_t)t * H + h], ex);
    }
}

// pass 3: scatter-add alpha-weighted messages.
// concat=1: out[dst, h*D+d] += alpha * xw[src, h*D+d]
// concat=0: out[dst, d]     += (alpha*scale) * xw[src, h*D+d]   (head mean)
__global__ void edge_agg_k(const int* __restrict__ src, const int* __restrict__ dst,
                           int E_, int n,
                           const float* __restrict__ xw,
                           const float* __restrict__ ealpha,
                           const float* __restrict__ denom,
                           float* __restrict__ outp,
                           int H, int D, int concat, float scale)
{
    long i = (long)blockIdx.x * blockDim.x + threadIdx.x;
    long total = ((long)E_ + n) * H;
    if (i >= total) return;
    long e = i / H;
    int  h = (int)(i % H);
    int s, t; edge_ids(e, src, dst, E_, s, t);
    float alpha = ealpha[e * H + h] / denom[(size_t)t * H + h];
    alpha *= scale;
    const float* row = xw + (size_t)s * H * D + (size_t)h * D;
    __builtin_prefetch(row, 0, 1);   // global_prefetch_b8 on gfx1250
    float* orow = concat ? (outp + (size_t)t * H * D + (size_t)h * D)
                         : (outp + (size_t)t * D);
    for (int d = 0; d < D; ++d) atomicAdd(&orow[d], alpha * row[d]);
}

// h = elu(agg + bias + lin)
__global__ void combine_elu_k(const float* __restrict__ agg,
                              const float* __restrict__ bias,
                              const float* __restrict__ lin,
                              float* __restrict__ hout, int n, int F)
{
    long i = (long)blockIdx.x * blockDim.x + threadIdx.x;
    if (i >= (long)n * F) return;
    int f = (int)(i % F);
    float v = agg[i] + bias[f] + lin[i];
    hout[i] = v > 0.f ? v : (__expf(v) - 1.0f);
}

__global__ void add_bias_k(float* __restrict__ outp, const float* __restrict__ b,
                           int n, int F)
{
    long i = (long)blockIdx.x * blockDim.x + threadIdx.x;
    if (i >= (long)n * F) return;
    outp[i] += b[i % F];
}

// ---------------------------------------------------------------------------
// Orchestration
// ---------------------------------------------------------------------------
static inline int cdiv_i(long a, long b) { return (int)((a + b - 1) / b); }

extern "C" void kernel_launch(void* const* d_in, const int* in_sizes, int n_in,
                              void* d_out, int out_size, void* d_ws, size_t ws_size,
                              hipStream_t stream)
{
    (void)in_sizes; (void)n_in; (void)out_size; (void)ws_size;
    const int N = GAT_N, E = GAT_E;
    const int EN = E + N;
    const int Mt = (N + 15) / 16;       // 6250 (exact)

    const float* x    = (const float*)d_in[0];
    const int*   ei   = (const int*)d_in[1];
    const int*   srcI = ei;
    const int*   dstI = ei + E;
    const float* W1  = (const float*)d_in[2];
    const float* as1 = (const float*)d_in[3];
    const float* ad1 = (const float*)d_in[4];
    const float* b1  = (const float*)d_in[5];
    const float* lw1 = (const float*)d_in[6];
    const float* lb1 = (const float*)d_in[7];
    const float* W2  = (const float*)d_in[8];
    const float* as2 = (const float*)d_in[9];
    const float* ad2 = (const float*)d_in[10];
    const float* b2  = (const float*)d_in[11];
    const float* lw2 = (const float*)d_in[12];
    const float* lb2 = (const float*)d_in[13];
    const float* W3  = (const float*)d_in[14];
    const float* as3 = (const float*)d_in[15];
    const float* ad3 = (const float*)d_in[16];
    const float* b3  = (const float*)d_in[17];
    const float* lw3 = (const float*)d_in[18];
    const float* lb3 = (const float*)d_in[19];
    float* out = (float*)d_out;

    // workspace carve-out (256B-aligned slabs)
    char* ws = (char*)d_ws;
    size_t off = 0;
    auto alloc_bytes = [&](size_t nbytes) {
        void* p = (void*)(ws + off);
        off = (off + nbytes + 255) & ~(size_t)255;
        return p;
    };
    auto allocf = [&](size_t nFloats) { return (float*)alloc_bytes(nFloats * 4); };
    auto alloch = [&](size_t nHalfs)  { return (_Float16*)alloc_bytes(nHalfs * 2); };

    float*    h1     = allocf((size_t)N * 40);
    float*    h2     = allocf((size_t)N * 40);
    float*    xw     = allocf((size_t)N * 726);          // max of 40 / 726
    float*    lin    = allocf((size_t)N * 40);
    float*    agg    = allocf((size_t)N * 40);
    float*    a_s    = allocf((size_t)N * 6);
    float*    a_d    = allocf((size_t)N * 6);
    unsigned* emax   = (unsigned*)allocf((size_t)N * 6);
    float*    denom  = allocf((size_t)N * 6);
    float*    ealpha = allocf((size_t)EN * 6);
    _Float16* Apack  = alloch((size_t)Mt * KT * 512);    // packed activations
    _Float16* Bpack  = alloch((size_t)KT * 46 * 512);    // packed weights (Nt<=46)

    const int TB = 256;
    const unsigned NEG_INF_ORD = 0x007FFFFFu;  // f2ord(-inf)

    auto pack_a = [&](const float* A, int K) {
        long tot = (long)Mt * KT * 512;
        pack_a_k<<<cdiv_i(tot, TB), TB, 0, stream>>>(A, Apack, N, K, Mt, KT);
    };
    auto gemm = [&](const float* Bmat, const float* bias, float* C, int K, int Nc) {
        int Nt = cdiv_i(Nc, 16);
        long tot = (long)KT * Nt * 512;
        pack_b_k<<<cdiv_i(tot, TB), TB, 0, stream>>>(Bmat, Bpack, K, Nc, KT, Nt);
        dim3 g(cdiv_i(Mt, 4), Nt);
        gemm_wmma_packed<<<g, dim3(32, 4), 0, stream>>>(
            (const v16h*)Apack, (const v16h*)Bpack, bias, C, Mt, Nt, N, Nc);
    };

    auto gat_edges = [&](const float* xw_, int H, int D, int concat, float scale,
                         float* target) {
        fill_u32<<<cdiv_i((long)N * H, TB), TB, 0, stream>>>(emax, NEG_INF_ORD, (long)N * H);
        fill_u32<<<cdiv_i((long)N * H, TB), TB, 0, stream>>>((unsigned*)denom, 0u, (long)N * H);
        edge_max_k<<<cdiv_i(EN, TB), TB, 0, stream>>>(srcI, dstI, E, N, a_s, a_d, emax, H);
        edge_expsum_k<<<cdiv_i(EN, TB), TB, 0, stream>>>(srcI, dstI, E, N, a_s, a_d,
                                                         emax, ealpha, denom, H);
        edge_agg_k<<<cdiv_i((long)EN * H, TB), TB, 0, stream>>>(srcI, dstI, E, N, xw_,
                                                                ealpha, denom, target,
                                                                H, D, concat, scale);
    };

    // ---------------- Layer 1: GATConv(50 -> 10, H=4, concat) + x@lw1 ----
    pack_a(x, 50);
    gemm(W1,  nullptr, xw,  50, 40);
    gemm(lw1, lb1,     lin, 50, 40);
    attn_scores<<<cdiv_i((long)N * 4, TB), TB, 0, stream>>>(xw, as1, ad1, a_s, a_d, N, 4, 10);
    fill_u32<<<cdiv_i((long)N * 40, TB), TB, 0, stream>>>((unsigned*)agg, 0u, (long)N * 40);
    gat_edges(xw, 4, 10, /*concat=*/1, 1.0f, agg);
    combine_elu_k<<<cdiv_i((long)N * 40, TB), TB, 0, stream>>>(agg, b1, lin, h1, N, 40);

    // ---------------- Layer 2: GATConv(40 -> 10, H=4, concat) + h@lw2 ----
    pack_a(h1, 40);
    gemm(W2,  nullptr, xw,  40, 40);
    gemm(lw2, lb2,     lin, 40, 40);
    attn_scores<<<cdiv_i((long)N * 4, TB), TB, 0, stream>>>(xw, as2, ad2, a_s, a_d, N, 4, 10);
    fill_u32<<<cdiv_i((long)N * 40, TB), TB, 0, stream>>>((unsigned*)agg, 0u, (long)N * 40);
    gat_edges(xw, 4, 10, /*concat=*/1, 1.0f, agg);
    combine_elu_k<<<cdiv_i((long)N * 40, TB), TB, 0, stream>>>(agg, b2, lin, h2, N, 40);

    // ------- Layer 3: GATConv(40 -> 121, H=6, mean) + h@lw3, no ELU ------
    pack_a(h2, 40);
    gemm(W3,  nullptr, xw,  40, 726);
    gemm(lw3, lb3,     out, 40, 121);               // out = h@lw3 + lb3 (overwrite)
    add_bias_k<<<cdiv_i((long)N * 121, TB), TB, 0, stream>>>(out, b3, N, 121);
    attn_scores<<<cdiv_i((long)N * 6, TB), TB, 0, stream>>>(xw, as3, ad3, a_s, a_d, N, 6, 121);
    gat_edges(xw, 6, 121, /*concat=*/0, 1.0f / 6.0f, out);  // mean over heads into out
}
